// GNN_v2_30932354465858
// MI455X (gfx1250) — matrix operand order repaired and checked by reading, not compiled, not run on hardware
//
#include <hip/hip_runtime.h>
#include <cstdint>

// ---------------------------------------------------------------------------
// GCN forward on MI455X (gfx1250, wave32).
//   layer1: agg1 = Anorm @ (x @ W1) ; h1 = relu(agg1 + b1)        F=128
//   layer2: agg2 = Anorm @ (h1 @ W2); h2 = relu(agg2 + b2)        F=64
//   head  : out  = log_softmax(h2 @ W3 + b3)                      F=16
//
// GEMMs: V_WMMA_F32_16X16X4_F32 (exact fp32; GEMMs are memory-bound so fp32
// WMMA costs nothing vs f16).  A-tiles (16 x K) are staged into LDS once per
// block by the Tensor Data Mover (tensor_load_to_lds + s_wait_tensorcnt),
// using the D# pad fields to insert 1 dword per K-dword row so WMMA fragment
// reads are bank-conflict-free.  Edge gather/scatter stays in the 192MB L2
// (feature matrix is 51MB) using wave-per-edge fp32 global atomics.
// ---------------------------------------------------------------------------

typedef __attribute__((ext_vector_type(2)))  float        v2f;
typedef __attribute__((ext_vector_type(8)))  float        v8f;
typedef __attribute__((ext_vector_type(16))) _Float16     v16h;
typedef __attribute__((ext_vector_type(4)))  unsigned int v4u;
typedef __attribute__((ext_vector_type(4)))  int          v4i;
typedef __attribute__((ext_vector_type(8)))  int          v8i;

#ifndef __has_builtin
#define __has_builtin(x) 0
#endif

#if __has_builtin(__builtin_amdgcn_wmma_f32_16x16x4_f32)
#define GCN_F32X4 1
#else
#define GCN_F32X4 0
#endif

#if __has_builtin(__builtin_amdgcn_tensor_load_to_lds) && \
    __has_builtin(__builtin_amdgcn_s_wait_tensorcnt)
#define GCN_TDM 1
#if __has_include(<hip/amd_detail/amd_gfx1250_TDM.h>)
#define GCN_TDM6 1 /* therock/clang-23: 6-arg builtin */
#else
#define GCN_TDM6 0 /* ROCm 7.2 / clang-22: 5-arg builtin */
#endif
#else
#define GCN_TDM 0
#endif

// ---------------- degree + symmetric normalization ----------------
__global__ void k_deg_init(float* __restrict__ deg, int n) {
  int i = blockIdx.x * blockDim.x + threadIdx.x;
  if (i < n) deg[i] = 1.0f;                       // self-loop weight
}

__global__ void k_deg_scatter(float* __restrict__ deg,
                              const long long* __restrict__ dst,
                              const float* __restrict__ ew, int e) {
  int i = blockIdx.x * blockDim.x + threadIdx.x;
  if (i < e) unsafeAtomicAdd(&deg[(size_t)dst[i]], ew[i]);
}

__global__ void k_dinv(const float* __restrict__ deg, float* __restrict__ dinv, int n) {
  int i = blockIdx.x * blockDim.x + threadIdx.x;
  if (i < n) {
    float d = deg[i];
    dinv[i] = d > 0.0f ? rsqrtf(d) : 0.0f;
  }
}

// ---------------- A-tile staging: TDM (async tensor DMA) into padded LDS ----
// Stages X[m0:m0+16, 0:K] into As with row stride K+1 dwords (pad via D#).
template <int K>
__device__ __forceinline__ void stage_tile(const float* __restrict__ X, int m0,
                                           float* As) {
  constexpr int KP = K + 1;
#if GCN_TDM
  if (threadIdx.y == 0) {                    // one wave issues the DMA
    unsigned long long ga = (unsigned long long)(uintptr_t)(X + (size_t)m0 * K);
    unsigned ldsb = (unsigned)(uintptr_t)As; // LDS byte address of tile
    // pad_interval code: #DWORDs before pad = 2<<code  (128 -> 6, 64 -> 5)
    constexpr unsigned padCode = (K == 128) ? 6u : 5u;

    v4u g0;
    g0[0] = 1u;                                            // count=1 (valid D#)
    g0[1] = ldsb;                                          // lds_addr
    g0[2] = (unsigned)(ga & 0xFFFFFFFFull);                // global_addr[31:0]
    g0[3] = (unsigned)((ga >> 32) & 0x01FFFFFFull)         // global_addr[56:32]
            | (2u << 30);                                  // type=2 ("image")

    const unsigned td0 = (unsigned)K;   // tensor_dim0 (elements per row)
    const unsigned td1 = 16u;           // tensor_dim1 (rows in view)
    v8i g1;
    g1[0] = (int)((2u << 16)            // data_size = 4 bytes
                  | (1u << 20)          // pad_enable
                  | (padCode << 22));   // pad_interval; pad_amount code 0 = 1 dword
    g1[1] = (int)((td0 & 0xFFFFu) << 16);                       // tensor_dim0 lo
    g1[2] = (int)((td0 >> 16) | ((td1 & 0xFFFFu) << 16));       // dim0 hi | dim1 lo
    g1[3] = (int)((td1 >> 16) | ((unsigned)K << 16));           // dim1 hi | tile_dim0=K
    g1[4] = (int)16;                                            // tile_dim1=16, tile_dim2=0
    g1[5] = (int)(unsigned)K;                                   // tensor_dim0_stride lo
    g1[6] = 0;                                                  // stride hi / dim1_stride lo
    g1[7] = 0;

    v4i gz = {0, 0, 0, 0};              // 2-D tile: groups 2/3 unused
#if GCN_TDM6
    v8i gz8 = {0, 0, 0, 0, 0, 0, 0, 0};
    __builtin_amdgcn_tensor_load_to_lds(g0, g1, gz, gz, gz8, 0);
#else
    __builtin_amdgcn_tensor_load_to_lds(g0, g1, gz, gz, 0);
#endif
    __builtin_amdgcn_s_wait_tensorcnt(0);
  }
#else
  // Fallback: cooperative coalesced copy with manual row padding.
  const int tid = threadIdx.y * 32 + threadIdx.x;
  const int nth = blockDim.y * 32;
  for (int i = tid; i < 16 * K; i += nth) {
    int r = i / K, c = i % K;   // K is a power of two
    As[r * KP + c] = X[(size_t)(m0 + r) * K + c];
  }
#endif
  __syncthreads();
}

// ---------------- WMMA GEMM: H[M x Nc] = X[M x K] @ W[K x Nc] ----------------
// block = (32, Nc/16): one wave per 16x16 output tile, one block per 16 rows.
template <int K>
__global__ void k_gemm_wmma(const float* __restrict__ X, const float* __restrict__ W,
                            float* __restrict__ H, int Nc) {
  constexpr int KP = K + 1;
  __shared__ float As[16 * KP];

  const int lane = threadIdx.x;          // 0..31
  const int grp  = lane >> 4;            // half-wave
  const int l    = lane & 15;
  const int m0   = blockIdx.x * 16;
  const int n0   = threadIdx.y * 16;

  stage_tile<K>(X, m0, As);

  v8f acc = {};
#if GCN_F32X4
  const float* ar = As + l * KP + grp * 2;
  for (int k0 = 0; k0 < K; k0 += 4) {
    v2f a, b;
    a.x = ar[k0];
    a.y = ar[k0 + 1];
    const float* wr = W + (size_t)(k0 + grp * 2) * Nc + n0 + l;
    b.x = wr[0];
    b.y = wr[Nc];
    acc = __builtin_amdgcn_wmma_f32_16x16x4_f32(false, a, false, b,
                                                (short)0, acc, false, false);
  }
#else
  const float* arow = As + l * KP;
  for (int kb = 0; kb < K; kb += 32) {
    v16h a, b;
    for (int j = 0; j < 16; ++j) {
      int ka = kb + 16 * grp + j;
      a[j] = (_Float16)arow[ka];
      b[j] = (_Float16)W[(size_t)(kb + 16 * grp + j) * Nc + n0 + l];
    }
    acc = __builtin_amdgcn_wmma_f32_16x16x32_f16(false, a, false, b,
                                                 (short)0, acc, false, false);
  }
#endif
  float* hr = H + (size_t)(m0 + grp * 8) * Nc + n0 + l;
  for (int i = 0; i < 8; ++i) hr[(size_t)i * Nc] = acc[i];
}

// ---------------- aggregation: self-loop init + edge scatter ----------------
__global__ void k_self_init(const float* __restrict__ h, const float* __restrict__ dinv,
                            float* __restrict__ agg, int total, int F) {
  int idx = blockIdx.x * blockDim.x + threadIdx.x;
  if (idx < total) {
    float d = dinv[idx / F];
    agg[idx] = h[idx] * d * d;       // self-loop: norm = dinv[i]*1*dinv[i]
  }
}

__global__ void k_edge_scatter(const float* __restrict__ h, const float* __restrict__ dinv,
                               const long long* __restrict__ src,
                               const long long* __restrict__ dst,
                               const float* __restrict__ ew,
                               float* __restrict__ agg, int E, int F) {
  int e    = (blockIdx.x * blockDim.x + threadIdx.x) >> 5;   // wave per edge
  int lane = threadIdx.x & 31;
  if (e >= E) return;
  long long s = src[e], d = dst[e];
  float nrm = dinv[s] * ew[e] * dinv[d];
  const float* hs = h + (size_t)s * F;
  float*       ad = agg + (size_t)d * F;
  for (int f = lane; f < F; f += 32)
    unsafeAtomicAdd(&ad[f], hs[f] * nrm);   // lands in L2 (h fits in 192MB)
}

__global__ void k_bias_relu(float* __restrict__ a, const float* __restrict__ b,
                            int total, int F) {
  int idx = blockIdx.x * blockDim.x + threadIdx.x;
  if (idx < total) a[idx] = fmaxf(a[idx] + b[idx & (F - 1)], 0.0f);  // F is pow2
}

// ---------------- head: WMMA GEMM (K=64, Nc=16) + fused log_softmax ----------
__global__ void k_head_logsoftmax(const float* __restrict__ X, const float* __restrict__ W,
                                  const float* __restrict__ bias,
                                  float* __restrict__ out) {
  constexpr int K  = 64;
  constexpr int KP = K + 1;
  constexpr int Nc = 16;
  __shared__ float As[16 * KP];

  const int lane = threadIdx.x;
  const int grp  = lane >> 4;
  const int l    = lane & 15;
  const int m0   = blockIdx.x * 16;

  stage_tile<K>(X, m0, As);

  v8f acc = {};
#if GCN_F32X4
  const float* ar = As + l * KP + grp * 2;
  for (int k0 = 0; k0 < K; k0 += 4) {
    v2f a, b;
    a.x = ar[k0];
    a.y = ar[k0 + 1];
    const float* wr = W + (size_t)(k0 + grp * 2) * Nc + l;
    b.x = wr[0];
    b.y = wr[Nc];
    acc = __builtin_amdgcn_wmma_f32_16x16x4_f32(false, a, false, b,
                                                (short)0, acc, false, false);
  }
#else
  const float* arow = As + l * KP;
  for (int kb = 0; kb < K; kb += 32) {
    v16h a, b;
    for (int j = 0; j < 16; ++j) {
      int ka = kb + 16 * grp + j;
      a[j] = (_Float16)arow[ka];
      b[j] = (_Float16)W[(size_t)(kb + 16 * grp + j) * Nc + l];
    }
    acc = __builtin_amdgcn_wmma_f32_16x16x32_f16(false, a, false, b,
                                                 (short)0, acc, false, false);
  }
#endif
  // C layout: lane l holds column n=l; row m = i + 8*grp for VGPR i.
  // Row reduction == reduction over the 16 lanes of each half-wave.
  float bn = bias[l];
  for (int i = 0; i < 8; ++i) {
    float v  = acc[i] + bn;
    float mx = v;
    for (int off = 8; off >= 1; off >>= 1) mx = fmaxf(mx, __shfl_xor(mx, off, 16));
    float ex = __expf(v - mx);
    float sm = ex;
    for (int off = 8; off >= 1; off >>= 1) sm += __shfl_xor(sm, off, 16);
    out[(size_t)(m0 + i + grp * 8) * 16 + l] = (v - mx) - __logf(sm);
  }
}

// ---------------------------------------------------------------------------
extern "C" void kernel_launch(void* const* d_in, const int* in_sizes, int n_in,
                              void* d_out, int out_size, void* d_ws, size_t ws_size,
                              hipStream_t stream) {
  (void)n_in; (void)out_size; (void)ws_size;
  const float*     x   = (const float*)d_in[0];
  const long long* ei  = (const long long*)d_in[1];   // int64 [2, E]
  const float*     ew  = (const float*)d_in[2];
  // d_in[3] (initial_x) unused by the reference
  const float*     W1  = (const float*)d_in[4];
  const float*     b1  = (const float*)d_in[5];
  const float*     W2  = (const float*)d_in[6];
  const float*     b2  = (const float*)d_in[7];
  const float*     W3  = (const float*)d_in[8];
  const float*     b3  = (const float*)d_in[9];

  const int HID = 128, MIDF = 64;
  const int N = in_sizes[0] / HID;     // 100000 (multiple of 16)
  const int E = in_sizes[2];           // 1000000
  const long long* src = ei;
  const long long* dst = ei + E;

  float* ws   = (float*)d_ws;
  float* deg  = ws;                         // N
  float* dinv = ws + N;                     // N
  float* h    = ws + 2 * (size_t)N;         // N*128 (per-layer GEMM output)
  float* agg  = h + (size_t)N * HID;        // N*128 (aggregation / h1)
  float* out  = (float*)d_out;

  const int T = 256;

  // degrees (with self-loops) -> dinv
  k_deg_init   <<<(N + T - 1) / T, T, 0, stream>>>(deg, N);
  k_deg_scatter<<<(E + T - 1) / T, T, 0, stream>>>(deg, dst, ew, E);
  k_dinv       <<<(N + T - 1) / T, T, 0, stream>>>(deg, dinv, N);

  // ---- layer 1 (F=128) ----
  k_gemm_wmma<128><<<dim3(N / 16), dim3(32, 8), 0, stream>>>(x, W1, h, HID);
  k_self_init   <<<(N * HID + T - 1) / T, T, 0, stream>>>(h, dinv, agg, N * HID, HID);
  k_edge_scatter<<<(E * 32 + T - 1) / T, T, 0, stream>>>(h, dinv, src, dst, ew, agg, E, HID);
  k_bias_relu   <<<(N * HID + T - 1) / T, T, 0, stream>>>(agg, b1, N * HID, HID);  // agg -> h1

  // ---- layer 2 (F=64): reuse halves of `h` ----
  float* h2pre = h;                          // N*64
  float* agg2  = h + (size_t)N * MIDF;       // N*64
  k_gemm_wmma<128><<<dim3(N / 16), dim3(32, 4), 0, stream>>>(agg, W2, h2pre, MIDF);
  k_self_init   <<<(N * MIDF + T - 1) / T, T, 0, stream>>>(h2pre, dinv, agg2, N * MIDF, MIDF);
  k_edge_scatter<<<(E * 32 + T - 1) / T, T, 0, stream>>>(h2pre, dinv, src, dst, ew, agg2, E, MIDF);
  k_bias_relu   <<<(N * MIDF + T - 1) / T, T, 0, stream>>>(agg2, b2, N * MIDF, MIDF); // agg2 -> h2

  // ---- head + log_softmax ----
  k_head_logsoftmax<<<dim3(N / 16), dim3(32), 0, stream>>>(agg2, W3, b3, out);
}
